// GNN_67456756351242
// MI455X (gfx1250) — matrix operand order repaired and checked by reading, not compiled, not run on hardware
//
#include <hip/hip_runtime.h>

typedef __attribute__((ext_vector_type(16))) _Float16 v16h;
typedef __attribute__((ext_vector_type(8)))  float    v8f;

#define HID 64
#define NEG_SLOPE 0.2f

__device__ __forceinline__ float leakyf(float e) { return e > 0.f ? e : NEG_SLOPE * e; }
// silu via hardware v_rcp_f32 (avoids the IEEE div_scale/div_fmas sequence)
__device__ __forceinline__ float siluf(float v)  {
  return v * __builtin_amdgcn_rcpf(1.f + __expf(-v));
}

// Order-preserving float <-> uint encoding for atomicMax on signed floats.
__device__ __forceinline__ unsigned enc_ord(float f) {
  unsigned u = __float_as_uint(f);
  return (u & 0x80000000u) ? ~u : (u | 0x80000000u);
}
__device__ __forceinline__ float dec_ord(unsigned u) {
  return (u & 0x80000000u) ? __uint_as_float(u & 0x7fffffffu) : __uint_as_float(~u);
}

// ---------------- setup kernels ----------------

__global__ void k_init_map(int* map, int N) {
  int i = blockIdx.x * blockDim.x + threadIdx.x;
  if (i < N) map[i] = -1;
}

__global__ void k_set_map(const long long* tgt, int* map, int T) {
  int t = blockIdx.x * blockDim.x + threadIdx.x;
  if (t < T) map[(int)tgt[t]] = t;   // duplicate targets: any winner is canonical
}

__global__ void k_init_slots(unsigned* smax, float* ssum, float* acc, int T) {
  int i = blockIdx.x * blockDim.x + threadIdx.x;
  if (i < T * HID) acc[i] = 0.f;
  if (i < T) { smax[i] = 0u; ssum[i] = 0.f; }   // enc_ord space: 0 < every real encoding
}

// ws = W @ att_src, wd = W @ att_dst   (a_src = x·ws since h = xW)
__global__ void k_wvec(const float* __restrict__ W, const float* __restrict__ att_src,
                       const float* __restrict__ att_dst, float* wsv, float* wdv) {
  int i = threadIdx.x;  // 64 threads
  float s = 0.f, d = 0.f;
  #pragma unroll 8
  for (int c = 0; c < HID; ++c) { float w = W[i * HID + c]; s += w * att_src[c]; d += w * att_dst[c]; }
  wsv[i] = s; wdv[i] = d;
}

__global__ void k_node_scores(const float* __restrict__ x, const float* __restrict__ wsv,
                              const float* __restrict__ wdv, float* asrc, float* adst, int N) {
  int n = blockIdx.x * blockDim.x + threadIdx.x;
  if (n >= N) return;
  const float4* xr = (const float4*)(x + (size_t)n * HID);
  const float4* a4 = (const float4*)wsv;
  const float4* b4 = (const float4*)wdv;
  float s = 0.f, d = 0.f;
  #pragma unroll
  for (int j = 0; j < 16; ++j) {
    float4 v = xr[j]; float4 a = a4[j]; float4 b = b4[j];
    s += v.x * a.x + v.y * a.y + v.z * a.z + v.w * a.w;
    d += v.x * b.x + v.y * b.y + v.z * b.z + v.w * b.w;
  }
  asrc[n] = s; adst[n] = d;
}

// ---------------- segment softmax over target destinations ----------------

__global__ void k_edge_max(const long long* __restrict__ src, const long long* __restrict__ dst,
                           const int* __restrict__ map, const float* __restrict__ asrc,
                           const float* __restrict__ adst, unsigned* smax, int E) {
  int i = blockIdx.x * blockDim.x + threadIdx.x;
  if (i >= E) return;
  // stream-ahead prefetch of the edge arrays (emits global_prefetch_b8)
  if (i + 8192 < E) {
    __builtin_prefetch(&dst[i + 8192], 0, 1);
    __builtin_prefetch(&src[i + 8192], 0, 1);
  }
  int d = (int)dst[i];
  int m = map[d];
  if (m < 0) return;                       // ~99% of edges filtered here
  int s = (int)src[i];
  float e = leakyf(asrc[s] + adst[d]);
  atomicMax(&smax[m], enc_ord(e));
}

__global__ void k_self_max(const long long* __restrict__ tgt, const int* __restrict__ map,
                           const float* __restrict__ asrc, const float* __restrict__ adst,
                           unsigned* smax, int T) {
  int t = blockIdx.x * blockDim.x + threadIdx.x;
  if (t >= T) return;
  int n = (int)tgt[t];
  if (map[n] != t) return;                 // process each unique node's self-loop once
  float e = leakyf(asrc[n] + adst[n]);
  atomicMax(&smax[t], enc_ord(e));
}

__global__ void k_edge_sum(const long long* __restrict__ src, const long long* __restrict__ dst,
                           const int* __restrict__ map, const float* __restrict__ asrc,
                           const float* __restrict__ adst, const unsigned* __restrict__ smax,
                           float* ssum, float* acc, const float* __restrict__ x, int E) {
  int i = blockIdx.x * blockDim.x + threadIdx.x;
  if (i >= E) return;
  if (i + 8192 < E) {
    __builtin_prefetch(&dst[i + 8192], 0, 1);
    __builtin_prefetch(&src[i + 8192], 0, 1);
  }
  int d = (int)dst[i];
  int m = map[d];
  if (m < 0) return;
  int s = (int)src[i];
  float e = leakyf(asrc[s] + adst[d]);
  float p = __expf(e - dec_ord(smax[m]));
  atomicAdd(&ssum[m], p);
  const float4* xr = (const float4*)(x + (size_t)s * HID);
  float* a = acc + (size_t)m * HID;
  #pragma unroll
  for (int j = 0; j < 16; ++j) {
    float4 v = xr[j];
    atomicAdd(&a[4 * j + 0], p * v.x);
    atomicAdd(&a[4 * j + 1], p * v.y);
    atomicAdd(&a[4 * j + 2], p * v.z);
    atomicAdd(&a[4 * j + 3], p * v.w);
  }
}

__global__ void k_self_sum(const long long* __restrict__ tgt, const int* __restrict__ map,
                           const float* __restrict__ asrc, const float* __restrict__ adst,
                           const unsigned* __restrict__ smax, float* ssum, float* acc,
                           const float* __restrict__ x, int T) {
  int t = blockIdx.x * blockDim.x + threadIdx.x;
  if (t >= T) return;
  int n = (int)tgt[t];
  if (map[n] != t) return;
  float e = leakyf(asrc[n] + adst[n]);
  float p = __expf(e - dec_ord(smax[t]));
  atomicAdd(&ssum[t], p);
  const float4* xr = (const float4*)(x + (size_t)n * HID);
  float* a = acc + (size_t)t * HID;
  #pragma unroll
  for (int j = 0; j < 16; ++j) {
    float4 v = xr[j];
    atomicAdd(&a[4 * j + 0], p * v.x);
    atomicAdd(&a[4 * j + 1], p * v.y);
    atomicAdd(&a[4 * j + 2], p * v.z);
    atomicAdd(&a[4 * j + 3], p * v.w);
  }
}

__global__ void k_scale_acc(float* acc, const float* __restrict__ ssum, int T) {
  int i = blockIdx.x * blockDim.x + threadIdx.x;
  if (i >= T * HID) return;
  acc[i] = acc[i] / (ssum[i / HID] + 1e-16f);
}

// ---------------- WMMA GEMM: C[M,64] = silu?(A[M,64] @ B[64,64] + bias) ----------------
// ISA 16-bit A 16x32 layout: lanes 0-15 row M=lane hold K {0..7,16..23}; lanes 16-31 hold K {8..15,24..31}
__device__ __forceinline__ v16h load_a16(const float* __restrict__ A, int M, int mbase,
                                         int koff, int l16, int half) {
  int row = mbase + l16; if (row >= M) row = M - 1;   // clamp (no effect when M%16==0)
  const float* r = A + (size_t)row * HID + koff + half * 8;
  float4 p0 = *(const float4*)(r + 0);
  float4 p1 = *(const float4*)(r + 4);
  float4 p2 = *(const float4*)(r + 16);
  float4 p3 = *(const float4*)(r + 20);
  v16h a;
  a[0] = (_Float16)p0.x; a[1] = (_Float16)p0.y; a[2]  = (_Float16)p0.z; a[3]  = (_Float16)p0.w;
  a[4] = (_Float16)p1.x; a[5] = (_Float16)p1.y; a[6]  = (_Float16)p1.z; a[7]  = (_Float16)p1.w;
  a[8] = (_Float16)p2.x; a[9] = (_Float16)p2.y; a[10] = (_Float16)p2.z; a[11] = (_Float16)p2.w;
  a[12] = (_Float16)p3.x; a[13] = (_Float16)p3.y; a[14] = (_Float16)p3.z; a[15] = (_Float16)p3.w;
  return a;
}

// B 32x16 (KxN): lanes 0-15 col N=lane hold K {0..7,16..23}; lanes 16-31 col N=lane-16 hold K {8..15,24..31}
__device__ __forceinline__ v16h load_b16(const float* __restrict__ B, int koff,
                                         int nbase, int l16, int half) {
  v16h b;
  const float* p = B + (size_t)(koff + half * 8) * HID + nbase + l16;
  #pragma unroll
  for (int e = 0; e < 16; ++e) b[e] = (_Float16)p[(e + (e >> 3) * 8) * HID];
  return b;
}

__global__ void k_gemm_k64_wmma(const float* __restrict__ A, const float* __restrict__ B,
                                const float* __restrict__ bias, float* __restrict__ C,
                                int M, int act_silu) {
  int mbase = blockIdx.x * 16, nbase = blockIdx.y * 16;
  int lane = threadIdx.x & 31, l16 = lane & 15, half = lane >> 4;
  v8f c = {};
  v16h a0 = load_a16(A, M, mbase, 0, l16, half);
  v16h b0 = load_b16(B, 0, nbase, l16, half);
  c = __builtin_amdgcn_wmma_f32_16x16x32_f16(false, a0, false, b0, (short)0, c, false, false);
  v16h a1 = load_a16(A, M, mbase, 32, l16, half);
  v16h b1 = load_b16(B, 32, nbase, l16, half);
  c = __builtin_amdgcn_wmma_f32_16x16x32_f16(false, a1, false, b1, (short)0, c, false, false);
  int n = nbase + l16;
  float bv = bias[n];                      // bias always provided: no branch
  // C layout: VGPR r -> M = mbase + r + 8*half, N = nbase + lane%16
  float* cp = C + (size_t)(mbase + half * 8) * HID + n;
  if (mbase + 16 <= M) {
    // full tile: single uniform branch, stores with immediate offsets (r*256B)
    #pragma unroll
    for (int r = 0; r < 8; ++r) {
      float v = c[r] + bv;
      if (act_silu) v = siluf(v);
      cp[r * HID] = v;
    }
  } else {
    #pragma unroll
    for (int r = 0; r < 8; ++r) {
      int m = mbase + half * 8 + r;
      if (m < M) {
        float v = c[r] + bv;
        if (act_silu) v = siluf(v);
        cp[r * HID] = v;
      }
    }
  }
}

// final: out[t] = feat2[map[tgt[t]]] . W2 + b2
__global__ void k_final(const long long* __restrict__ tgt, const int* __restrict__ map,
                        const float* __restrict__ feat2, const float* __restrict__ W2,
                        const float* __restrict__ b2, float* out, int T) {
  int t = blockIdx.x * blockDim.x + threadIdx.x;
  if (t >= T) return;
  int m = map[(int)tgt[t]]; if (m < 0) m = 0;
  const float4* f = (const float4*)(feat2 + (size_t)m * HID);
  const float4* w = (const float4*)W2;
  float acc = b2[0];
  #pragma unroll
  for (int j = 0; j < 16; ++j) {
    float4 a = f[j]; float4 b = w[j];
    acc += a.x * b.x + a.y * b.y + a.z * b.z + a.w * b.w;
  }
  out[t] = acc;
}

// ---------------- host launch ----------------

extern "C" void kernel_launch(void* const* d_in, const int* in_sizes, int n_in,
                              void* d_out, int out_size, void* d_ws, size_t ws_size,
                              hipStream_t stream) {
  const float*     x       = (const float*)d_in[0];
  const long long* ei      = (const long long*)d_in[1];
  const long long* tgt     = (const long long*)d_in[2];
  const float*     W       = (const float*)d_in[3];
  const float*     att_src = (const float*)d_in[4];
  const float*     att_dst = (const float*)d_in[5];
  const float*     bias    = (const float*)d_in[6];
  const float*     W1      = (const float*)d_in[7];
  const float*     b1      = (const float*)d_in[8];
  const float*     W2      = (const float*)d_in[9];
  const float*     b2      = (const float*)d_in[10];

  const int N = in_sizes[0] / HID;       // 100000
  const int E = in_sizes[1] / 2;         // 1600000
  const int T = in_sizes[2];             // 1024
  const long long* src = ei;
  const long long* dst = ei + E;

  char* w = (char*)d_ws;
  auto alloc = [&](size_t bytes) -> char* {
    char* p = w; w += (bytes + 255) & ~(size_t)255; return p;
  };
  float*    asrc  = (float*)alloc((size_t)N * 4);
  float*    adst  = (float*)alloc((size_t)N * 4);
  int*      map   = (int*)alloc((size_t)N * 4);
  float*    wsv   = (float*)alloc(HID * 4);
  float*    wdv   = (float*)alloc(HID * 4);
  unsigned* smax  = (unsigned*)alloc((size_t)T * 4);
  float*    ssum  = (float*)alloc((size_t)T * 4);
  float*    acc   = (float*)alloc((size_t)T * HID * 4);
  float*    feat1 = (float*)alloc((size_t)T * HID * 4);
  float*    feat2 = (float*)alloc((size_t)T * HID * 4);

  k_init_map  <<<(N + 255) / 256, 256, 0, stream>>>(map, N);
  k_set_map   <<<(T + 255) / 256, 256, 0, stream>>>(tgt, map, T);
  k_init_slots<<<(T * HID + 255) / 256, 256, 0, stream>>>(smax, ssum, acc, T);
  k_wvec      <<<1, HID, 0, stream>>>(W, att_src, att_dst, wsv, wdv);
  k_node_scores<<<(N + 255) / 256, 256, 0, stream>>>(x, wsv, wdv, asrc, adst, N);

  k_edge_max<<<(E + 255) / 256, 256, 0, stream>>>(src, dst, map, asrc, adst, smax, E);
  k_self_max<<<(T + 255) / 256, 256, 0, stream>>>(tgt, map, asrc, adst, smax, T);
  k_edge_sum<<<(E + 255) / 256, 256, 0, stream>>>(src, dst, map, asrc, adst, smax, ssum, acc, x, E);
  k_self_sum<<<(T + 255) / 256, 256, 0, stream>>>(tgt, map, asrc, adst, smax, ssum, acc, x, T);
  k_scale_acc<<<(T * HID + 255) / 256, 256, 0, stream>>>(acc, ssum, T);

  dim3 gg((T + 15) / 16, HID / 16);
  k_gemm_k64_wmma<<<gg, 32, 0, stream>>>(acc,   W,  bias, feat1, T, 1);  // silu(gat_out)
  k_gemm_k64_wmma<<<gg, 32, 0, stream>>>(feat1, W1, b1,   feat2, T, 1);  // silu(@W1+b1)

  k_final<<<(T + 255) / 256, 256, 0, stream>>>(tgt, map, feat2, W2, b2, (float*)d_out, T);
}
